// attention_48833778156182
// MI455X (gfx1250) — matrix operand order repaired and checked by reading, not compiled, not run on hardware
//
#include <hip/hip_runtime.h>

#define DIM_IN 512
#define DIM_K  256
#define DIM_V  256
#define NB     4
#define SEQ    4096

typedef __attribute__((ext_vector_type(16))) __bf16       v16bf;
typedef __attribute__((ext_vector_type(8)))  float        v8f;
typedef __attribute__((ext_vector_type(4)))  unsigned int v4u;
typedef __attribute__((ext_vector_type(8)))  unsigned int v8u;
typedef __attribute__((ext_vector_type(4)))  int          v4i;

// Address-space-qualified int4 for the async global->LDS builtin
typedef __attribute__((address_space(1))) v4i as1_v4i;
typedef __attribute__((address_space(3))) v4i as3_v4i;

#if defined(__has_builtin)
#if __has_builtin(__builtin_amdgcn_global_load_async_to_lds_b128)
#define HAVE_ASYNC_LDS 1
#endif
#endif
#ifndef HAVE_ASYNC_LDS
#define HAVE_ASYNC_LDS 0
#endif

static __device__ __forceinline__ unsigned short f2bf_bits(float f) {
    unsigned u = __builtin_bit_cast(unsigned, f);
    unsigned r = u + 0x7FFFu + ((u >> 16) & 1u);   // round-to-nearest-even
    return (unsigned short)(r >> 16);
}

// Load 16 bf16 (two 16B chunks) into a WMMA operand register block.
static __device__ __forceinline__ v16bf ld2g(const unsigned short* p0,
                                             const unsigned short* p1) {
    v8u t;
    t.lo = *(const v4u*)p0;
    t.hi = *(const v4u*)p1;
    return __builtin_bit_cast(v16bf, t);
}

static __device__ __forceinline__ v8f wmma_bf16(v16bf a, v16bf b, v8f c) {
    return __builtin_amdgcn_wmma_f32_16x16x32_bf16(
        /*neg_a=*/false, a, /*neg_b=*/false, b,
        /*c_mod=*/(short)0, c, /*reuse_a=*/false, /*reuse_b=*/false);
}

static __device__ __forceinline__ v8f zero8() {
    v8f z = {0.f, 0.f, 0.f, 0.f, 0.f, 0.f, 0.f, 0.f};
    return z;
}

// 16-byte global -> LDS copy: async path if available, else via registers.
static __device__ __forceinline__ void cp16(unsigned short* lds,
                                            const unsigned short* g) {
#if HAVE_ASYNC_LDS
    __builtin_amdgcn_global_load_async_to_lds_b128(
        (as1_v4i*)(unsigned long long)(const void*)g,
        (as3_v4i*)(unsigned)(unsigned long long)(void*)lds, 0, 0);
#else
    *(v4u*)lds = *(const v4u*)g;
#endif
}

static __device__ __forceinline__ void wait_async_copies() {
#if HAVE_ASYNC_LDS
#if __has_builtin(__builtin_amdgcn_s_wait_asynccnt)
    __builtin_amdgcn_s_wait_asynccnt(0);
#else
    asm volatile("s_wait_asynccnt 0x0" ::: "memory");
#endif
#endif
}

// ---------------------------------------------------------------- conversions
__global__ void cvt_x_kernel(const float* __restrict__ x,
                             unsigned short* __restrict__ xb, int n) {
    int i = blockIdx.x * blockDim.x + threadIdx.x;
    if (i < n) xb[i] = f2bf_bits(x[i]);
}

// W[k][n] (512x256) fp32 -> Wt[n][k] (256x512) bf16, for 3 matrices
__global__ void cvt_w_kernel(const float* __restrict__ Wq,
                             const float* __restrict__ Wk,
                             const float* __restrict__ Wv,
                             unsigned short* __restrict__ Wtq,
                             unsigned short* __restrict__ Wtk,
                             unsigned short* __restrict__ Wtv, int n) {
    int i = blockIdx.x * blockDim.x + threadIdx.x;
    if (i >= n) return;
    int z   = i / (DIM_IN * DIM_K);
    int rem = i % (DIM_IN * DIM_K);
    int col = rem / DIM_IN;  // output feature n
    int k   = rem % DIM_IN;
    const float*    W  = (z == 0) ? Wq : (z == 1) ? Wk : Wv;
    unsigned short* Wt = (z == 0) ? Wtq : (z == 1) ? Wtk : Wtv;
    Wt[(size_t)col * DIM_IN + k] = f2bf_bits(W[(size_t)k * DIM_K + col]);
}

// ---------------------------------------------------------- QKV projection
// One wave per 16x16 output tile. z = 0:Q 1:K 2:V(transposed store).
__global__ __launch_bounds__(32) void qkv_gemm_kernel(
    const unsigned short* __restrict__ Xbf,
    const unsigned short* __restrict__ Wtq,
    const unsigned short* __restrict__ Wtk,
    const unsigned short* __restrict__ Wtv,
    const float* __restrict__ bq, const float* __restrict__ bk,
    const float* __restrict__ bv,
    unsigned short* __restrict__ Qb, unsigned short* __restrict__ Kb,
    unsigned short* __restrict__ Vt) {
    const int lane = threadIdx.x;
    const int ln   = lane & 15;
    const int hf   = lane >> 4;
    const int row0 = blockIdx.x * 16;
    const int col0 = blockIdx.y * 16;
    const int z    = blockIdx.z;

    const unsigned short* Wt   = (z == 0) ? Wtq : (z == 1) ? Wtk : Wtv;
    const float*          bias = (z == 0) ? bq : (z == 1) ? bk : bv;

    const unsigned short* ar = Xbf + (size_t)(row0 + ln) * DIM_IN;
    const unsigned short* br = Wt + (size_t)(col0 + ln) * DIM_IN;

    v8f acc = zero8();
#pragma unroll
    for (int kc = 0; kc < DIM_IN / 32; ++kc) {
        const int ka = kc * 32 + hf * 8;   // A lane K base
        const int kb = kc * 32 + hf * 8;   // B lane K base
        v16bf a = ld2g(ar + ka, ar + ka + 16);
        v16bf b = ld2g(br + kb, br + kb + 16);
        acc = wmma_bf16(a, b, acc);
    }

    const float bcol = bias[col0 + ln];
    if (z < 2) {
        unsigned short* Out = (z == 0) ? Qb : Kb;
#pragma unroll
        for (int r = 0; r < 8; ++r) {
            int row = row0 + r + hf * 8;
            Out[(size_t)row * DIM_K + col0 + ln] = f2bf_bits(acc[r] + bcol);
        }
    } else {  // V: store transposed Vt[b][dim][seq]
#pragma unroll
        for (int r = 0; r < 8; ++r) {
            int srow = row0 + r + hf * 8;
            int bidx = srow >> 12;
            int s    = srow & (SEQ - 1);
            Vt[((size_t)(bidx * DIM_V + col0 + ln)) * SEQ + s] =
                f2bf_bits(acc[r] + bcol);
        }
    }
}

// --------------------------------------------------------- flash attention
// 4 waves per block (64 queries); K/V tiles (32 keys) staged once into LDS
// per iteration and shared by all 4 waves => 4x less L2 traffic.
#define KPAD 264  // 32 K-rows, 256 bf16 + 8 pad  => 528B stride (conflict-free)
#define VPAD 40   // 256 V-rows, 32 bf16 + 8 pad  =>  80B stride (conflict-free)

__global__ __launch_bounds__(128) void flash_attn_kernel(
    const unsigned short* __restrict__ Qb,
    const unsigned short* __restrict__ Kb,
    const unsigned short* __restrict__ Vt,
    const int* __restrict__ mask, float* __restrict__ out) {
    const int tid  = threadIdx.x;     // 0..127
    const int w    = tid >> 5;        // wave id 0..3
    const int lane = tid & 31;
    const int ln   = lane & 15;
    const int hf   = lane >> 4;
    const int q0   = blockIdx.x * 64 + w * 16;
    const int b    = blockIdx.y;
    const float scale = 0.0625f;  // 1/sqrt(256)

    __shared__ __align__(16) unsigned short KtL[32 * KPAD];   // [key][dim]
    __shared__ __align__(16) unsigned short VlL[256 * VPAD];  // [dim][key]
    __shared__ __align__(16) float          Sf[4][16][34];
    __shared__ __align__(16) unsigned short Pl[4][16][32];
    __shared__ float mrow[4][16], lrow[4][16], arow[4][16];

    const unsigned short* qrow = Qb + ((size_t)(b * SEQ + q0 + ln)) * DIM_K;

    v8f o[16];
#pragma unroll
    for (int g = 0; g < 16; ++g) o[g] = zero8();

    if (lane < 16) { mrow[w][lane] = -1e30f; lrow[w][lane] = 0.f; }
    __syncthreads();

    for (int j = 0; j < SEQ / 32; ++j) {
        const int key0 = j * 32;

        // ---- cooperative staging of K (32x256) and V (256x32) tiles to LDS
#pragma unroll
        for (int i = 0; i < 8; ++i) {
            int id  = tid + i * 128;        // 0..1023
            int row = id >> 5;              // key row 0..31
            int cc  = (id & 31) * 8;        // dim chunk
            cp16(&KtL[row * KPAD + cc],
                 Kb + ((size_t)(b * SEQ + key0 + row)) * DIM_K + cc);
        }
#pragma unroll
        for (int i = 0; i < 8; ++i) {
            int id  = tid + i * 128;        // 0..1023
            int dim = id >> 2;              // 0..255
            int cc  = (id & 3) * 8;         // key chunk 0/8/16/24
            cp16(&VlL[dim * VPAD + cc],
                 Vt + ((size_t)(b * DIM_V + dim)) * SEQ + key0 + cc);
        }
        if (j + 1 < SEQ / 32) {  // nudge next K tile toward L2/L0
            __builtin_prefetch(
                Kb + ((size_t)(b * SEQ + key0 + 32 + lane)) * DIM_K, 0, 1);
        }
        wait_async_copies();
        __syncthreads();

        // ---- S = Q x K^T  (two 16x16 tiles, B operands from LDS)
        v8f s0 = zero8(), s1 = zero8();
#pragma unroll
        for (int c = 0; c < 8; ++c) {
            const int ka = c * 32 + hf * 8;
            const int kb = c * 32 + hf * 8;
            v16bf a  = ld2g(qrow + ka, qrow + ka + 16);
            const unsigned short* p0 = &KtL[ln * KPAD + kb];
            const unsigned short* p1 = &KtL[(16 + ln) * KPAD + kb];
            v16bf b0 = ld2g(p0, p0 + 16);
            v16bf b1 = ld2g(p1, p1 + 16);
            s0 = wmma_bf16(a, b0, s0);
            s1 = wmma_bf16(a, b1, s1);
        }

        // ---- scale + key-padding mask, spill score tile to LDS (fp32)
        const bool ma = mask[b * SEQ + key0 + ln] != 0;
        const bool mb = mask[b * SEQ + key0 + 16 + ln] != 0;
#pragma unroll
        for (int r = 0; r < 8; ++r) {
            s0[r] = ma ? s0[r] * scale : -1e9f;
            s1[r] = mb ? s1[r] * scale : -1e9f;
        }
#pragma unroll
        for (int r = 0; r < 8; ++r) {
            int row = r + hf * 8;
            Sf[w][row][ln]      = s0[r];
            Sf[w][row][16 + ln] = s1[r];
        }
        __syncthreads();

        // ---- online softmax (lane t handles row t), P -> LDS bf16
        if (lane < 16) {
            const int row = lane;
            float mo = mrow[w][row], mx = mo;
#pragma unroll
            for (int c2 = 0; c2 < 32; ++c2) mx = fmaxf(mx, Sf[w][row][c2]);
            float al  = __expf(mo - mx);
            float sum = 0.f;
#pragma unroll
            for (int c2 = 0; c2 < 32; ++c2) {
                float p = __expf(Sf[w][row][c2] - mx);
                sum += p;
                Pl[w][row][c2] = f2bf_bits(p);
            }
            lrow[w][row] = lrow[w][row] * al + sum;
            mrow[w][row] = mx;
            arow[w][row] = al;
        }
        __syncthreads();

        // ---- rescale O accumulators by alpha[row]
        float alr[8];
#pragma unroll
        for (int r = 0; r < 8; ++r) alr[r] = arow[w][r + hf * 8];
#pragma unroll
        for (int g = 0; g < 16; ++g)
#pragma unroll
            for (int r = 0; r < 8; ++r) o[g][r] *= alr[r];

        // ---- P fragment (A operand, 16x32) from LDS
        const unsigned short* pp = &Pl[w][ln][hf * 8];
        v16bf pf = ld2g(pp, pp + 16);

        // ---- O += P x V  (V B-operands from LDS: lane=dim, K=keys)
#pragma unroll
        for (int g = 0; g < 16; ++g) {
            const unsigned short* vp = &VlL[(g * 16 + ln) * VPAD + hf * 8];
            v16bf vf = ld2g(vp, vp + 16);
            o[g] = wmma_bf16(pf, vf, o[g]);
        }
        __syncthreads();  // before next iteration overwrites KtL/VlL
    }

    // ---- normalize and store fp32 output
    float invl[8];
#pragma unroll
    for (int r = 0; r < 8; ++r) invl[r] = 1.f / lrow[w][r + hf * 8];
#pragma unroll
    for (int g = 0; g < 16; ++g)
#pragma unroll
        for (int r = 0; r < 8; ++r) {
            int row = q0 + r + hf * 8;
            out[((size_t)(b * SEQ + row)) * DIM_V + g * 16 + ln] =
                o[g][r] * invl[r];
        }
}

// ------------------------------------------------------------------- launch
extern "C" void kernel_launch(void* const* d_in, const int* in_sizes, int n_in,
                              void* d_out, int out_size, void* d_ws,
                              size_t ws_size, hipStream_t stream) {
    const float* x    = (const float*)d_in[0];
    const int*   mask = (const int*)d_in[1];
    const float* Wq   = (const float*)d_in[2];
    const float* bq   = (const float*)d_in[3];
    const float* Wk   = (const float*)d_in[4];
    const float* bk   = (const float*)d_in[5];
    const float* Wv   = (const float*)d_in[6];
    const float* bv   = (const float*)d_in[7];
    float*       out  = (float*)d_out;

    unsigned short* ws = (unsigned short*)d_ws;
    size_t off = 0;
    unsigned short* Xbf = ws + off; off += (size_t)NB * SEQ * DIM_IN;
    unsigned short* Wtq = ws + off; off += (size_t)DIM_K * DIM_IN;
    unsigned short* Wtk = ws + off; off += (size_t)DIM_K * DIM_IN;
    unsigned short* Wtv = ws + off; off += (size_t)DIM_V * DIM_IN;
    unsigned short* Qb  = ws + off; off += (size_t)NB * SEQ * DIM_K;
    unsigned short* Kb  = ws + off; off += (size_t)NB * SEQ * DIM_K;
    unsigned short* Vt  = ws + off; off += (size_t)NB * SEQ * DIM_V;

    const int nx = NB * SEQ * DIM_IN;
    cvt_x_kernel<<<(nx + 255) / 256, 256, 0, stream>>>(x, Xbf, nx);

    const int nw = 3 * DIM_IN * DIM_K;
    cvt_w_kernel<<<(nw + 255) / 256, 256, 0, stream>>>(Wq, Wk, Wv, Wtq, Wtk,
                                                       Wtv, nw);

    qkv_gemm_kernel<<<dim3(NB * SEQ / 16, DIM_K / 16, 3), 32, 0, stream>>>(
        Xbf, Wtq, Wtk, Wtv, bq, bk, bv, Qb, Kb, Vt);

    flash_attn_kernel<<<dim3(SEQ / 64, NB), 128, 0, stream>>>(Qb, Kb, Vt, mask,
                                                              out);
}
